// SynSelfAtt_6975026889313
// MI455X (gfx1250) — compile-verified, matrix operand order checked by reading, not run on hardware
//
#include <hip/hip_runtime.h>
#include <hip/hip_bf16.h>

typedef __attribute__((ext_vector_type(16))) _Float16 v16h;
typedef __attribute__((ext_vector_type(8)))  _Float16 v8h;
typedef __attribute__((ext_vector_type(8)))  float    v8f;
typedef __attribute__((ext_vector_type(4)))  float    f32x4;

#define IN_DIM   256
#define H_DIM    256
#define NUM_NODES 32768
#define MAXN     16
#define RPN      17            // rows per node (1 center + 16 neigh)
#define NPW      32            // nodes per workgroup
#define ROWS     (NPW * RPN)   // 544
#define MT_A     (ROWS / 16)   // 34 M-tiles in phase A
#define KT       (IN_DIM / 32) // 8 K-tiles of 32 for f16 WMMA

static __device__ __forceinline__ float sigm(float x) {
    return 1.0f / (1.0f + __expf(-x));
}
static __device__ __forceinline__ float fast_tanh(float x) {
    return 1.0f - 2.0f / (__expf(2.0f * x) + 1.0f);
}

static __device__ __forceinline__ v8f wmma_f16(v16h a, v16h b, v8f c) {
    return __builtin_amdgcn_wmma_f32_16x16x32_f16(
        /*neg_a=*/false, a, /*neg_b=*/false, b,
        /*c_mod=*/(short)0, c, /*reuse_a=*/false, /*reuse_b=*/false);
}

// A/B fragment from a row-major f16 matrix (global): lane = row (A: M, B: N col
// of W since B[k][n] = W[n][k]); K split per ISA: lanes 0-15 -> K {0..7,16..23},
// lanes 16-31 -> K {8..15,24..31}. Two 16B loads per lane.
static __device__ __forceinline__ v16h load_frag_g16(const _Float16* __restrict__ base,
                                                     int ld, int row0, int k0, int lane) {
    int r  = row0 + (lane & 15);
    int kk = k0 + ((lane >> 4) << 3);
    const _Float16* p = base + (size_t)r * ld + kk;
    union { v16h v; v8h h[2]; } u;
    u.h[0] = *(const v8h*)(p);
    u.h[1] = *(const v8h*)(p + 16);
    return u.v;
}

// Same fragment pattern but from LDS, with caller-supplied per-lane row index
// (lets us gather strided center rows / per-node neighbor rows for free).
static __device__ __forceinline__ v16h load_frag_lds(const _Float16* base,
                                                     int rowIdx, int k0, int lane) {
    int kk = k0 + ((lane >> 4) << 3);
    const _Float16* p = base + rowIdx * H_DIM + kk;
    union { v16h v; v8h h[2]; } u;
    u.h[0] = *(const v8h*)(p);
    u.h[1] = *(const v8h*)(p + 16);
    return u.v;
}

// A fragment from fp32 x: 2x 32B per lane, converted to f16 in-register.
static __device__ __forceinline__ v16h load_frag_x(const float* __restrict__ xrow0,
                                                   int k0, int lane) {
    const float* p = xrow0 + (size_t)(lane & 15) * IN_DIM + k0 + ((lane >> 4) << 3);
    f32x4 a0 = *(const f32x4*)(p);
    f32x4 a1 = *(const f32x4*)(p + 4);
    f32x4 b0 = *(const f32x4*)(p + 16);
    f32x4 b1 = *(const f32x4*)(p + 20);
    v16h r;
#pragma unroll
    for (int i = 0; i < 4; ++i) {
        r[i]      = (_Float16)a0[i];
        r[4 + i]  = (_Float16)a1[i];
        r[8 + i]  = (_Float16)b0[i];
        r[12 + i] = (_Float16)b1[i];
    }
    return r;
}

// Per-(node, 16-col tile): sigmoid -> masked softmax over k=16 (8 k's per lane
// half, combined via shfl_xor 16) -> con = sum_k a*neigh. Writes conL f16.
static __device__ __forceinline__ void softmax_con(
    v8f t, int n, int g, float bias, int nvalid, int kb, int lane,
    const _Float16* projL, const _Float16* hcL, _Float16* conL) {
    float hc = (float)hcL[n * H_DIM + g];
    float s[8];
    float mx = -1e30f;
#pragma unroll
    for (int v = 0; v < 8; ++v) {
        float sv = sigm(t[v] + hc + bias);
        s[v] = ((kb + v) < nvalid) ? sv : -1e30f;
        mx = fmaxf(mx, s[v]);
    }
    mx = fmaxf(mx, __shfl_xor(mx, 16, 32));
    float sum = 0.0f;
#pragma unroll
    for (int v = 0; v < 8; ++v) {
        float e = ((kb + v) < nvalid) ? __expf(s[v] - mx) : 0.0f;
        s[v] = e;
        sum += e;
    }
    sum += __shfl_xor(sum, 16, 32);
    float inv = 1.0f / sum;
    float con = 0.0f;
#pragma unroll
    for (int v = 0; v < 8; ++v) {
        float nh = (float)projL[(n * RPN + 1 + kb + v) * H_DIM + g];
        con += s[v] * inv * nh;
    }
    con += __shfl_xor(con, 16, 32);
    if (lane < 16) conL[n * H_DIM + g] = (_Float16)con;
}

__global__ void convert_weights(const float* __restrict__ w0, const float* __restrict__ w1,
                                const float* __restrict__ w2, const float* __restrict__ w3,
                                const float* __restrict__ w4, _Float16* __restrict__ dst) {
    int i = blockIdx.x * blockDim.x + threadIdx.x;
    if (i < H_DIM * H_DIM) {
        dst[i]                    = (_Float16)w0[i];
        dst[1 * H_DIM * H_DIM + i] = (_Float16)w1[i];
        dst[2 * H_DIM * H_DIM + i] = (_Float16)w2[i];
        dst[3 * H_DIM * H_DIM + i] = (_Float16)w3[i];
        dst[4 * H_DIM * H_DIM + i] = (_Float16)w4[i];
    }
}

__global__ __launch_bounds__(256) void synsa_main(
    const float* __restrict__ x, const int* __restrict__ counts,
    const _Float16* __restrict__ wx, const _Float16* __restrict__ wh,
    const _Float16* __restrict__ uh, const _Float16* __restrict__ wg,
    const _Float16* __restrict__ ug,
    const float* __restrict__ bx, const float* __restrict__ bh,
    const float* __restrict__ buh, const float* __restrict__ bg,
    const float* __restrict__ bug,
    float* __restrict__ out) {
    extern __shared__ _Float16 smem[];
    _Float16* projL = smem;                    // [544][256] f16 = 278.5 KB
    _Float16* hcL   = projL + ROWS * H_DIM;    // [32][256]  f16 = 16 KB
    _Float16* conL  = hcL + NPW * H_DIM;       // [32][256]  f16 = 16 KB

    const int tid   = threadIdx.x;
    const int lane  = tid & 31;
    const int wave  = tid >> 5;
    const int g0    = wave * 32;           // 32-col N-chunk per wave
    const int col   = lane & 15;
    const int rbase = (lane >> 4) << 3;    // C-fragment row base (0 or 8)
    const int node0 = blockIdx.x * NPW;
    const int gA = g0 + col;
    const int gB = g0 + 16 + col;
    const float* xblk = x + (size_t)node0 * RPN * IN_DIM;

    // ---------------- Phase A: proj = tanh(X * Wx^T + bx) ----------------
    float bx0 = bx[gA], bx1 = bx[gB];
    v16h BW0[KT], BW1[KT];                  // Wx fragments register-resident
#pragma unroll
    for (int kt = 0; kt < KT; ++kt) {
        BW0[kt] = load_frag_g16(wx, IN_DIM, g0,      kt * 32, lane);
        BW1[kt] = load_frag_g16(wx, IN_DIM, g0 + 16, kt * 32, lane);
    }
    for (int mt = 0; mt < MT_A; ++mt) {
        v8f c0 = {}, c1 = {};
        const float* xrow = xblk + (size_t)mt * 16 * IN_DIM;
#pragma unroll
        for (int kt = 0; kt < KT; ++kt) {
            v16h a = load_frag_x(xrow, kt * 32, lane);
            c0 = wmma_f16(a, BW0[kt], c0);
            c1 = wmma_f16(a, BW1[kt], c1);
        }
        int rw = mt * 16 + rbase;
#pragma unroll
        for (int v = 0; v < 8; ++v) {
            projL[(rw + v) * H_DIM + gA] = (_Float16)fast_tanh(c0[v] + bx0);
            projL[(rw + v) * H_DIM + gB] = (_Float16)fast_tanh(c1[v] + bx1);
        }
    }
    __syncthreads();

    // ------------- Phase B1: hc = center @ Wh^T (wave-local cols) -------------
    for (int mt = 0; mt < 2; ++mt) {
        v8f h0 = {}, h1 = {};
#pragma unroll
        for (int kt = 0; kt < KT; ++kt) {
            v16h a  = load_frag_lds(projL, (mt * 16 + col) * RPN, kt * 32, lane);
            v16h b0 = load_frag_g16(wh, H_DIM, g0,      kt * 32, lane);
            v16h b1 = load_frag_g16(wh, H_DIM, g0 + 16, kt * 32, lane);
            h0 = wmma_f16(a, b0, h0);
            h1 = wmma_f16(a, b1, h1);
        }
        int nb = mt * 16 + rbase;
#pragma unroll
        for (int v = 0; v < 8; ++v) {
            hcL[(nb + v) * H_DIM + gA] = (_Float16)h0[v];
            hcL[(nb + v) * H_DIM + gB] = (_Float16)h1[v];
        }
    }

    // ------ Phase B2: per-node Uh GEMM + masked softmax + weighted sum ------
    float bt0 = bh[gA] + buh[gA];
    float bt1 = bh[gB] + buh[gB];
    v16h BU0[KT], BU1[KT];                  // Uh fragments register-resident
#pragma unroll
    for (int kt = 0; kt < KT; ++kt) {
        BU0[kt] = load_frag_g16(uh, H_DIM, g0,      kt * 32, lane);
        BU1[kt] = load_frag_g16(uh, H_DIM, g0 + 16, kt * 32, lane);
    }
    for (int n = 0; n < NPW; ++n) {
        int nvalid = counts[node0 + n] + 1;   // 1..16 valid neighbors
        v8f t0 = {}, t1 = {};
#pragma unroll
        for (int kt = 0; kt < KT; ++kt) {
            v16h a = load_frag_lds(projL, n * RPN + 1 + col, kt * 32, lane);
            t0 = wmma_f16(a, BU0[kt], t0);
            t1 = wmma_f16(a, BU1[kt], t1);
        }
        softmax_con(t0, n, gA, bt0, nvalid, rbase, lane, projL, hcL, conL);
        softmax_con(t1, n, gB, bt1, nvalid, rbase, lane, projL, hcL, conL);
    }
    __syncthreads();

    // -------- Phase B3: gate = sigmoid(C*Wg^T + Con*Ug^T + b); blend --------
    float bgg0 = bg[gA] + bug[gA];
    float bgg1 = bg[gB] + bug[gB];
    for (int mt = 0; mt < 2; ++mt) {
        v8f q0 = {}, q1 = {};
#pragma unroll
        for (int kt = 0; kt < KT; ++kt) {
            v16h ac = load_frag_lds(projL, (mt * 16 + col) * RPN, kt * 32, lane);
            v16h an = load_frag_lds(conL,  mt * 16 + col,         kt * 32, lane);
            v16h b0 = load_frag_g16(wg, H_DIM, g0,      kt * 32, lane);
            v16h b1 = load_frag_g16(wg, H_DIM, g0 + 16, kt * 32, lane);
            v16h d0 = load_frag_g16(ug, H_DIM, g0,      kt * 32, lane);
            v16h d1 = load_frag_g16(ug, H_DIM, g0 + 16, kt * 32, lane);
            q0 = wmma_f16(ac, b0, q0); q0 = wmma_f16(an, d0, q0);
            q1 = wmma_f16(ac, b1, q1); q1 = wmma_f16(an, d1, q1);
        }
        int nb = mt * 16 + rbase;
#pragma unroll
        for (int v = 0; v < 8; ++v) {
            int n = nb + v;
            float gate0 = sigm(q0[v] + bgg0);
            float gate1 = sigm(q1[v] + bgg1);
            float cen0  = (float)projL[n * RPN * H_DIM + gA];
            float cen1  = (float)projL[n * RPN * H_DIM + gB];
            float cc0   = (float)conL[n * H_DIM + gA];
            float cc1   = (float)conL[n * H_DIM + gB];
            out[(size_t)(node0 + n) * H_DIM + gA] = gate0 * cen0 + (1.0f - gate0) * cc0;
            out[(size_t)(node0 + n) * H_DIM + gB] = gate1 * cen1 + (1.0f - gate1) * cc1;
        }
    }
}

extern "C" void kernel_launch(void* const* d_in, const int* in_sizes, int n_in,
                              void* d_out, int out_size, void* d_ws, size_t ws_size,
                              hipStream_t stream) {
    const float* x     = (const float*)d_in[0];
    const int*   counts = (const int*)d_in[1];
    const float* Wx_w = (const float*)d_in[2];  const float* Wx_b = (const float*)d_in[3];
    const float* Wh_w = (const float*)d_in[4];  const float* Wh_b = (const float*)d_in[5];
    const float* Uh_w = (const float*)d_in[6];  const float* Uh_b = (const float*)d_in[7];
    const float* Wg_w = (const float*)d_in[8];  const float* Wg_b = (const float*)d_in[9];
    const float* Ug_w = (const float*)d_in[10]; const float* Ug_b = (const float*)d_in[11];

    _Float16* w16 = (_Float16*)d_ws;           // 5 x 256x256 f16 = 640 KB scratch
    convert_weights<<<(H_DIM * H_DIM + 255) / 256, 256, 0, stream>>>(
        Wx_w, Wh_w, Uh_w, Wg_w, Ug_w, w16);

    size_t ldsBytes = (size_t)(ROWS + 2 * NPW) * H_DIM * sizeof(_Float16); // 311296 B
    (void)hipFuncSetAttribute((const void*)synsa_main,
                              hipFuncAttributeMaxDynamicSharedMemorySize,
                              (int)ldsBytes);
    const int W = H_DIM * H_DIM;
    synsa_main<<<NUM_NODES / NPW, 256, ldsBytes, stream>>>(
        x, counts,
        w16, w16 + W, w16 + 2 * W, w16 + 3 * W, w16 + 4 * W,
        Wx_b, Wh_b, Uh_b, Wg_b, Ug_b,
        (float*)d_out);
}
